// GATGINEncoder_11570641895556
// MI455X (gfx1250) — matrix operand order repaired and checked by reading, not compiled, not run on hardware
//
#include <hip/hip_runtime.h>

#define DD 128

#ifndef __has_builtin
#define __has_builtin(x) 0
#endif

typedef __attribute__((ext_vector_type(16))) __bf16 v16bf;
typedef __attribute__((ext_vector_type(8)))  float  v8f;

// exact parameter type for the async-LDS builtin (from hipcc diagnostic):
// int __attribute__((vector_size(16))) in addrspace(1)/(3)
typedef int v4i __attribute__((__vector_size__(4 * sizeof(int))));
typedef __attribute__((address_space(1))) void  gvoid;
typedef __attribute__((address_space(3))) void  lvoid;
typedef __attribute__((address_space(1))) v4i   gv4i;
typedef __attribute__((address_space(3))) v4i   lv4i;

// Async global->LDS staging only on the device pass, and only if the gfx1250
// builtin exists in this toolchain; otherwise fall back to manual staging.
#if defined(__HIP_DEVICE_COMPILE__) && __has_builtin(__builtin_amdgcn_global_load_async_to_lds_b128)
#define USE_ASYNC_LDS 1
#else
#define USE_ASYNC_LDS 0
#endif

__device__ __forceinline__ unsigned short f32_to_bf16_bits(float f){
  __bf16 b = (__bf16)f;                       // native v_cvt (RNE)
  return __builtin_bit_cast(unsigned short, b);
}
// order-preserving float<->uint for atomicMax-based segment max
__device__ __forceinline__ unsigned flip_f32(float f){
  unsigned u = __float_as_uint(f);
  return (u & 0x80000000u) ? ~u : (u | 0x80000000u);
}
__device__ __forceinline__ float unflip_f32(unsigned u){
  unsigned v = (u & 0x80000000u) ? (u & 0x7fffffffu) : ~u;
  return __uint_as_float(v);
}

__global__ void zero_kernel(float* __restrict__ p, size_t n){
  size_t i = (size_t)blockIdx.x * blockDim.x + threadIdx.x;
  const size_t stride = (size_t)gridDim.x * blockDim.x;
  for (; i < n; i += stride) p[i] = 0.f;
}

// W[k][n] f32 (row-major over k)  ->  Wt[n][k] bf16 (k contiguous, pair-loadable)
__global__ void pack_weight_kernel(const float* __restrict__ W,
                                   unsigned short* __restrict__ out){
  const int idx = blockIdx.x * blockDim.x + threadIdx.x;
  if (idx >= DD * DD) return;
  const int nn = idx >> 7;
  const int k  = idx & (DD - 1);
  out[idx] = f32_to_bf16_bits(W[k * DD + nn]);
}

// ---------------------------------------------------------------------------
// 16x128-tile GEMM: out[row, n] = epilogue( (A(+A2)) @ Wt^T + bias )
// MODE 0: store (A@W + bias)                    -> x_l / x_r projections
// MODE 1: A+=A2; BN((A@W)+b); ReLU; store       -> GIN hidden
// MODE 2: ReLU((A@W)+b); atomicAdd to pool[batch[row]]  -> final GEMM + pool
// One wave32 per 16x16 WMMA tile, K=128 in 4 steps of 32 (bf16, f32 accum).
// A tile staged via GLOBAL_LOAD_ASYNC_TO_LDS_B128 (ASYNCcnt) when available.
// ---------------------------------------------------------------------------
template<int MODE>
__global__ __launch_bounds__(256) void gemm128_kernel(
    const float* __restrict__ A, const float* __restrict__ A2,
    const unsigned short* __restrict__ Wt, const float* __restrict__ bias,
    const float* __restrict__ bn_gamma, const float* __restrict__ bn_beta,
    const float* __restrict__ bn_mean,  const float* __restrict__ bn_var,
    const int* __restrict__ batch, float* __restrict__ out, int nrows)
{
  __shared__ float sA [16 * DD];
  __shared__ float sA2[(MODE == 1) ? 16 * DD : 16];
  const int row0 = blockIdx.x << 4;
  const int t = threadIdx.x;

  { // stage 16x128 f32 activation rows into LDS (32B per thread)
    const int r = t >> 4;
    const int c = (t & 15) << 3;
    const size_t gidx = (size_t)(row0 + r) * DD + c;
    const bool inb = (row0 + r) < nrows;
#if USE_ASYNC_LDS
    if (inb) {
      __builtin_amdgcn_global_load_async_to_lds_b128(
          (gv4i*)(gvoid*)(A + gidx),     (lv4i*)(lvoid*)&sA[r * DD + c],     0, 0);
      __builtin_amdgcn_global_load_async_to_lds_b128(
          (gv4i*)(gvoid*)(A + gidx + 4), (lv4i*)(lvoid*)&sA[r * DD + c + 4], 0, 0);
      if (MODE == 1) {
        __builtin_amdgcn_global_load_async_to_lds_b128(
            (gv4i*)(gvoid*)(A2 + gidx),     (lv4i*)(lvoid*)&sA2[r * DD + c],     0, 0);
        __builtin_amdgcn_global_load_async_to_lds_b128(
            (gv4i*)(gvoid*)(A2 + gidx + 4), (lv4i*)(lvoid*)&sA2[r * DD + c + 4], 0, 0);
      }
    } else {
#pragma unroll
      for (int i = 0; i < 8; i++) {
        sA[r * DD + c + i] = 0.f;
        if (MODE == 1) sA2[r * DD + c + i] = 0.f;
      }
    }
#if __has_builtin(__builtin_amdgcn_s_wait_asynccnt)
    __builtin_amdgcn_s_wait_asynccnt(0);
#else
    asm volatile("s_wait_asynccnt 0" ::: "memory");
#endif
#else
    float v[8];
    if (inb) {
#pragma unroll
      for (int i = 0; i < 8; i++) v[i] = A[gidx + i];
    } else {
#pragma unroll
      for (int i = 0; i < 8; i++) v[i] = 0.f;
    }
#pragma unroll
    for (int i = 0; i < 8; i++) sA[r * DD + c + i] = v[i];
    if (MODE == 1) {
      if (inb) {
#pragma unroll
        for (int i = 0; i < 8; i++) v[i] = A2[gidx + i];
      } else {
#pragma unroll
        for (int i = 0; i < 8; i++) v[i] = 0.f;
      }
#pragma unroll
      for (int i = 0; i < 8; i++) sA2[r * DD + c + i] = v[i];
    }
#endif
  }
  __syncthreads();

  const int wave = t >> 5;        // 8 waves -> 8 column tiles of 16
  const int lane = t & 31;
  const int g    = lane >> 4;     // half-wave group
  const int ln   = lane & 15;
  const int n    = (wave << 4) + ln;

  const unsigned short* wcol = Wt + (size_t)n * DD;
  __builtin_prefetch(wcol, 0, 1);             // global_prefetch_b8 on weights

  v8f acc = {0.f, 0.f, 0.f, 0.f, 0.f, 0.f, 0.f, 0.f};

#pragma unroll
  for (int kt = 0; kt < 4; kt++) {
    const int kbase = kt * 32;
    union { __bf16 h[16]; unsigned u32[8]; v16bf v; } af, bfu;
    // A fragment: ISA 16-bit A 16x32 layout (K pairs per VGPR, +8g / +16 split)
#pragma unroll
    for (int vv = 0; vv < 8; vv++) {
      const int k = kbase + ((vv < 4) ? (8 * g + 2 * vv)
                                      : (16 + 8 * g + 2 * (vv - 4)));
      float2 p = *(const float2*)&sA[ln * DD + k];
      if (MODE == 1) {
        const float2 q = *(const float2*)&sA2[ln * DD + k];
        p.x += q.x; p.y += q.y;
      }
      af.h[2 * vv]     = (__bf16)p.x;
      af.h[2 * vv + 1] = (__bf16)p.y;
    }
    // B fragment: column n per lane; packed bf16 K-pair loads drop straight in
#pragma unroll
    for (int vv = 0; vv < 8; vv++) {
      const int k = kbase + ((vv < 4) ? (8 * g + 2 * vv)
                                      : (16 + 8 * g + 2 * (vv - 4)));
      bfu.u32[vv] = *(const unsigned*)(wcol + k);
    }
    acc = __builtin_amdgcn_wmma_f32_16x16x32_bf16(false, af.v, false, bfu.v,
                                                  (short)0, acc, false, false);
  }

  const float bv = bias[n];
  float scale = 1.f, shift = 0.f;
  if (MODE == 1) {
    const float inv = rsqrtf(bn_var[n] + 1e-5f);
    scale = inv * bn_gamma[n];
    shift = bn_beta[n] - bn_mean[n] * scale;
  }
#pragma unroll
  for (int r = 0; r < 8; r++) {   // C/D layout: row = r + 8g, col = n
    const int row = row0 + r + (g << 3);
    if (row < nrows) {
      float val = acc[r] + bv;
      if (MODE == 0) {
        out[(size_t)row * DD + n] = val;
      } else if (MODE == 1) {
        val = fmaxf(val * scale + shift, 0.f);
        out[(size_t)row * DD + n] = val;
      } else {
        val = fmaxf(val, 0.f);    // relu; leaky_relu of relu == relu
        atomicAdd(&out[(size_t)batch[row] * DD + n], val);
      }
    }
  }
}

// ---- GATv2 edge passes (wave per edge; self-loops are virtual edges E..E+N) -

__global__ __launch_bounds__(256) void edge_logits_kernel(
    const float* __restrict__ xl, const float* __restrict__ xr,
    const int* __restrict__ ei, const float* __restrict__ att,
    float* __restrict__ logits, unsigned* __restrict__ menc, int E, int Nn)
{
  const int w    = (int)(((size_t)blockIdx.x * blockDim.x + threadIdx.x) >> 5);
  const int lane = threadIdx.x & 31;
  if (w >= E + Nn) return;
  int src, dst;
  if (w < E) { src = ei[w]; dst = ei[E + w]; } else { src = w - E; dst = src; }
  const int c = lane << 2;                       // 4 channels / lane
  const float* pa = xl + (size_t)src * DD + c;
  const float* pb = xr + (size_t)dst * DD + c;
  const float* pw = att + c;                     // att flat index == channel
  float part = 0.f;
#pragma unroll
  for (int i = 0; i < 4; i++) {
    float s = pa[i] + pb[i];
    s = (s > 0.f) ? s : 0.2f * s;                // leaky_relu 0.2
    part += s * pw[i];
  }
  // reduce 8 lanes (one head = 32 channels) -> head logit
  part += __shfl_xor(part, 1, 32);
  part += __shfl_xor(part, 2, 32);
  part += __shfl_xor(part, 4, 32);
  if ((lane & 7) == 0) {
    const int h = lane >> 3;
    logits[(size_t)w * 4 + h] = part;
    atomicMax(&menc[(size_t)dst * 4 + h], flip_f32(part));
  }
}

__global__ __launch_bounds__(256) void softmax_num_kernel(
    float* __restrict__ logits, const unsigned* __restrict__ menc,
    float* __restrict__ den, const int* __restrict__ ei, int E, int Nn)
{
  const int e = blockIdx.x * blockDim.x + threadIdx.x;
  if (e >= E + Nn) return;
  const int dst = (e < E) ? ei[E + e] : (e - E);
#pragma unroll
  for (int h = 0; h < 4; h++) {
    const float m   = unflip_f32(menc[(size_t)dst * 4 + h]);
    const float num = __expf(logits[(size_t)e * 4 + h] - m);
    logits[(size_t)e * 4 + h] = num;             // overwrite with numerator
    atomicAdd(&den[(size_t)dst * 4 + h], num);
  }
}

__global__ __launch_bounds__(256) void gat_scatter_kernel(
    const float* __restrict__ xl, const float* __restrict__ logits,
    const float* __restrict__ den, const int* __restrict__ ei,
    float* __restrict__ acc, int E, int Nn)
{
  const int w    = (int)(((size_t)blockIdx.x * blockDim.x + threadIdx.x) >> 5);
  const int lane = threadIdx.x & 31;
  if (w >= E + Nn) return;
  int src, dst;
  if (w < E) { src = ei[w]; dst = ei[E + w]; } else { src = w - E; dst = src; }
  const int h = lane >> 3;
  const float alpha = logits[(size_t)w * 4 + h] / den[(size_t)dst * 4 + h];
  const int c = lane << 2;
  const float* ps = xl + (size_t)src * DD + c;
  float* pd = acc + (size_t)dst * DD + c;
#pragma unroll
  for (int i = 0; i < 4; i++) atomicAdd(&pd[i], alpha * ps[i]);
}

__global__ void gat_bias_act_kernel(float* __restrict__ acc,
                                    const float* __restrict__ gb, size_t total){
  size_t i = (size_t)blockIdx.x * blockDim.x + threadIdx.x;
  const size_t stride = (size_t)gridDim.x * blockDim.x;
  for (; i < total; i += stride) {
    float v = acc[i] + gb[i & (DD - 1)];
    acc[i] = (v > 0.f) ? v : 0.01f * v;          // leaky_relu 0.01
  }
}

__global__ __launch_bounds__(256) void gin_scatter_kernel(
    const float* __restrict__ x1, const int* __restrict__ ei,
    float* __restrict__ agg, int E)
{
  const int w    = (int)(((size_t)blockIdx.x * blockDim.x + threadIdx.x) >> 5);
  const int lane = threadIdx.x & 31;
  if (w >= E) return;
  const int src = ei[w];
  const int dst = ei[E + w];
  const int c = lane << 2;
  const float* ps = x1 + (size_t)src * DD + c;
  float* pd = agg + (size_t)dst * DD + c;
#pragma unroll
  for (int i = 0; i < 4; i++) atomicAdd(&pd[i], ps[i]);
}

extern "C" void kernel_launch(void* const* d_in, const int* in_sizes, int n_in,
                              void* d_out, int out_size, void* d_ws, size_t ws_size,
                              hipStream_t stream)
{
  const float* x        = (const float*)d_in[0];
  const int*   ei       = (const int*)  d_in[1];
  const int*   batch    = (const int*)  d_in[2];
  const float* W_l      = (const float*)d_in[3];
  const float* b_l      = (const float*)d_in[4];
  const float* W_r      = (const float*)d_in[5];
  const float* b_r      = (const float*)d_in[6];
  const float* att      = (const float*)d_in[7];
  const float* gat_bias = (const float*)d_in[8];
  const float* W1       = (const float*)d_in[9];
  const float* b1       = (const float*)d_in[10];
  const float* bn_gamma = (const float*)d_in[11];
  const float* bn_beta  = (const float*)d_in[12];
  const float* bn_mean  = (const float*)d_in[13];
  const float* bn_var   = (const float*)d_in[14];
  const float* W2       = (const float*)d_in[15];
  const float* b2       = (const float*)d_in[16];
  float* out = (float*)d_out;

  const int N     = in_sizes[0] / DD;
  const int E     = in_sizes[1] / 2;
  const int total = E + N;

  // workspace carve-out (buffers aliased across phases)
  char* base = (char*)d_ws;
  size_t off = 0;
  auto take = [&](size_t bytes) -> char* {
    char* p = base + off;
    off = (off + bytes + 255) & ~(size_t)255;
    return p;
  };
  float*          xl     = (float*)take((size_t)N * DD * 4);   // later: h
  float*          xr     = (float*)take((size_t)N * DD * 4);   // later: agg
  float*          acc    = (float*)take((size_t)N * DD * 4);   // becomes x1
  float*          logits = (float*)take((size_t)total * 4 * 4);
  float*          den    = (float*)take((size_t)N * 4 * 4);
  unsigned*       menc   = (unsigned*)take((size_t)N * 4 * 4);
  unsigned short* wp     = (unsigned short*)take((size_t)4 * DD * DD * 2);
  (void)ws_size; (void)n_in;
  float* h   = xl;   // alias: xl dead after gat_scatter
  float* agg = xr;   // alias: xr dead after edge_logits

  const int ZT = 256;
  auto zb = [&](size_t n) -> int {
    size_t b = (n + ZT - 1) / ZT;
    if (b > 65535) b = 65535;   // grid-stride covers remainder
    return (int)b;
  };
  zero_kernel<<<zb((size_t)N * DD), ZT, 0, stream>>>(acc, (size_t)N * DD);
  zero_kernel<<<zb((size_t)N * 4),  ZT, 0, stream>>>(den, (size_t)N * 4);
  zero_kernel<<<zb((size_t)N * 4),  ZT, 0, stream>>>((float*)menc, (size_t)N * 4);
  zero_kernel<<<zb((size_t)out_size), ZT, 0, stream>>>(out, (size_t)out_size);

  pack_weight_kernel<<<(DD * DD + 255) / 256, 256, 0, stream>>>(W_l, wp + 0 * DD * DD);
  pack_weight_kernel<<<(DD * DD + 255) / 256, 256, 0, stream>>>(W_r, wp + 1 * DD * DD);
  pack_weight_kernel<<<(DD * DD + 255) / 256, 256, 0, stream>>>(W1,  wp + 2 * DD * DD);
  pack_weight_kernel<<<(DD * DD + 255) / 256, 256, 0, stream>>>(W2,  wp + 3 * DD * DD);

  const int gemm_blocks = (N + 15) / 16;
  gemm128_kernel<0><<<gemm_blocks, 256, 0, stream>>>(
      x, nullptr, wp + 0 * DD * DD, b_l,
      nullptr, nullptr, nullptr, nullptr, nullptr, xl, N);
  gemm128_kernel<0><<<gemm_blocks, 256, 0, stream>>>(
      x, nullptr, wp + 1 * DD * DD, b_r,
      nullptr, nullptr, nullptr, nullptr, nullptr, xr, N);

  const int eblocks = (total + 7) / 8;   // 8 wave32 edges per 256-thread block
  edge_logits_kernel<<<eblocks, 256, 0, stream>>>(xl, xr, ei, att, logits, menc, E, N);

  zero_kernel<<<zb((size_t)N * DD), ZT, 0, stream>>>(agg, (size_t)N * DD);

  softmax_num_kernel<<<(total + 255) / 256, 256, 0, stream>>>(logits, menc, den, ei, E, N);
  gat_scatter_kernel<<<eblocks, 256, 0, stream>>>(xl, logits, den, ei, acc, E, N);
  gat_bias_act_kernel<<<zb((size_t)N * DD), 256, 0, stream>>>(acc, gat_bias, (size_t)N * DD);
  gin_scatter_kernel<<<(E + 7) / 8, 256, 0, stream>>>(acc, ei, agg, E);

  gemm128_kernel<1><<<gemm_blocks, 256, 0, stream>>>(
      acc, agg, wp + 2 * DD * DD, b1,
      bn_gamma, bn_beta, bn_mean, bn_var, nullptr, h, N);
  gemm128_kernel<2><<<gemm_blocks, 256, 0, stream>>>(
      h, nullptr, wp + 3 * DD * DD, b2,
      nullptr, nullptr, nullptr, nullptr, batch, out, N);
}
